// MixtureOfExperts_34522947125507
// MI455X (gfx1250) — compile-verified, matrix operand order checked by reading, not run on hardware
//
#include <hip/hip_runtime.h>
#include <hip/hip_bf16.h>

// ---------------------------------------------------------------------------
// MoE top-2 (H=1024, E=8, FF=4096, T=4096) for MI455X / gfx1250.
// Sparse routing (2/8 experts) + bf16 WMMA GEMMs (v_wmma_f32_16x16x32_bf16)
// with Tensor Data Mover (tensor_load_to_lds) double-buffered tile DMA.
// Workspace requirement: ~230 MB.
// ---------------------------------------------------------------------------

#define T_   4096
#define H_   1024
#define E_   8
#define FF_  4096
#define A_   (T_ * 2)          // 8192 routed assignments (top-2)
#define RP_  9216              // padded row capacity = 72 tiles of 128
#define BM_  128
#define BN_  128
#define BK_  32
#define LDT_ 40                // LDS tile row stride in halves (32 + 8 pad, 80B)

typedef __attribute__((ext_vector_type(16))) __bf16 v16bf;
typedef __attribute__((ext_vector_type(8)))  float  v8f;
typedef __attribute__((ext_vector_type(4)))  unsigned int uint32x4_t;
typedef __attribute__((ext_vector_type(8)))  int          int32x8_t;
typedef __attribute__((ext_vector_type(4)))  int          int32x4_t;

union FragBF { unsigned int u[8]; v16bf v; };

__device__ __forceinline__ unsigned short f2bf(float f) {
  unsigned int u = __float_as_uint(f);
  u = (u + 0x7FFFu + ((u >> 16) & 1u)) >> 16;   // round-to-nearest-even
  return (unsigned short)u;
}

// ---------------------------------------------------------------------------
// TDM: DMA one [128 rows x 32 halves] bf16 tile, global -> LDS, padding each
// 16-DWORD row with 4 DWORDs so the LDS row stride is LDT_=40 halves.
// D# packing per CDNA5 ISA ch.8 (group0: count/lds/global/type, group1: dims).
// Issue from one wave only; completion via s_wait_tensorcnt.
// ---------------------------------------------------------------------------
__device__ __forceinline__ void tdm_load_tile(unsigned int lds_addr,
                                              const unsigned short* gptr,
                                              int kstride /* elements */) {
  unsigned long long ga = (unsigned long long)(uintptr_t)gptr;
  uint32x4_t g0;
  g0.x = 1u;                                           // count=1 (valid), user mode
  g0.y = lds_addr;                                     // LDS byte address
  g0.z = (unsigned int)(ga & 0xFFFFFFFFull);           // global_addr[31:0]
  g0.w = (unsigned int)((ga >> 32) & 0x01FFFFFFull)    // global_addr[56:32]
         | (2u << 30);                                 // type = 2 ("image")
  int32x8_t g1;
  g1[0] = (int)((1u << 16)     // data_size = 1 -> 2 bytes
              | (1u << 20)     // pad_enable
              | (3u << 22)     // pad_interval code 3 -> every 16 DWORDs (one row)
              | (3u << 25));   // pad_amount code 3 -> 4 DWORDs (8 halves)
  g1[1] = (int)(((unsigned)kstride & 0xFFFFu) << 16);  // tensor_dim0 lo16
  g1[2] = (int)(((unsigned)kstride >> 16) & 0xFFFFu);  // tensor_dim0 hi16 | dim1 lo16=0
  g1[3] = (int)(16u | ((unsigned)BK_ << 16));          // tensor_dim1 = 1<<20, tile_dim0 = 32
  g1[4] = (int)((unsigned)BM_);                        // tile_dim1 = 128, tile_dim2 = 0
  g1[5] = (int)(unsigned)kstride;                      // tensor_dim0_stride lo32
  g1[6] = 0;                                           // stride0 hi16 | dim1_stride lo16
  g1[7] = 0;                                           // dim1_stride hi32 (unused, 2D)
  int32x4_t gz = {0, 0, 0, 0};
#if __clang_major__ >= 23
  int32x8_t gz8 = {0, 0, 0, 0, 0, 0, 0, 0};
  __builtin_amdgcn_tensor_load_to_lds(g0, g1, gz, gz, gz8, 0);
#else
  __builtin_amdgcn_tensor_load_to_lds(g0, g1, gz, gz, 0);
#endif
}

// ---------------------------------------------------------------------------
// 1) init: zero output accumulator, counts; tok_id = -1 everywhere
// ---------------------------------------------------------------------------
__global__ void moe_init_kernel(float* __restrict__ out, int* __restrict__ tok_id,
                                unsigned int* __restrict__ counts) {
  int idx    = blockIdx.x * blockDim.x + threadIdx.x;
  int stride = gridDim.x * blockDim.x;
  for (int i = idx; i < T_ * H_; i += stride) out[i] = 0.0f;
  for (int i = idx; i < RP_;     i += stride) tok_id[i] = -1;
  if (idx < E_) counts[idx] = 0u;
}

// ---------------------------------------------------------------------------
// 2) gating: one wave32 per token. Wg staged in LDS (32 KB), lane-strided dot,
//    wave reduction via shfl_xor, softmax -> top-2 -> renormalize.
// ---------------------------------------------------------------------------
__global__ void __launch_bounds__(256)
moe_gate_kernel(const float* __restrict__ x, const float* __restrict__ Wg,
                const float* __restrict__ bg, unsigned int* __restrict__ counts,
                int* __restrict__ asg_tok, int* __restrict__ asg_exp,
                float* __restrict__ asg_w) {
  __shared__ float sWg[H_ * E_];
  for (int i = threadIdx.x; i < H_ * E_; i += 256) sWg[i] = Wg[i];
  __syncthreads();

  int wid  = threadIdx.x >> 5;
  int lane = threadIdx.x & 31;
  int t    = blockIdx.x * 8 + wid;           // grid.x = T_/8

  float acc[E_];
#pragma unroll
  for (int e = 0; e < E_; ++e) acc[e] = 0.0f;

  const float* xr = x + (size_t)t * H_;
  for (int h = lane; h < H_; h += 32) {
    float xv = xr[h];
#pragma unroll
    for (int e = 0; e < E_; ++e) acc[e] += xv * sWg[h * E_ + e];
  }
#pragma unroll
  for (int e = 0; e < E_; ++e)
    for (int off = 16; off >= 1; off >>= 1)
      acc[e] += __shfl_xor(acc[e], off, 32);

  if (lane == 0) {
    float mx = -3.0e38f;
#pragma unroll
    for (int e = 0; e < E_; ++e) { acc[e] += bg[e]; mx = fmaxf(mx, acc[e]); }
    float p[E_];
#pragma unroll
    for (int e = 0; e < E_; ++e) p[e] = __expf(acc[e] - mx);
    // top-2 (ties -> lowest index, matching jax.lax.top_k)
    int i1 = 0;
#pragma unroll
    for (int e = 1; e < E_; ++e) if (p[e] > p[i1]) i1 = e;
    int i2 = (i1 == 0) ? 1 : 0;
#pragma unroll
    for (int e = 0; e < E_; ++e) if (e != i1 && p[e] > p[i2]) i2 = e;
    float s  = p[i1] + p[i2];
    float w1 = p[i1] / s, w2 = p[i2] / s;    // renormalized (softmax den cancels)
    asg_tok[t * 2 + 0] = t;  asg_exp[t * 2 + 0] = i1;  asg_w[t * 2 + 0] = w1;
    asg_tok[t * 2 + 1] = t;  asg_exp[t * 2 + 1] = i2;  asg_w[t * 2 + 1] = w2;
    atomicAdd(&counts[i1], 1u);
    atomicAdd(&counts[i2], 1u);
  }
}

// ---------------------------------------------------------------------------
// 3) offsets: pad each expert segment to a multiple of BM_ so every 128-row
//    GEMM tile belongs to exactly one expert; fill row -> expert map.
// ---------------------------------------------------------------------------
__global__ void moe_offsets_kernel(const unsigned int* __restrict__ counts,
                                   unsigned int* __restrict__ offsets,
                                   unsigned int* __restrict__ cursors,
                                   int* __restrict__ expert_row) {
  __shared__ unsigned int soff[E_ + 1];
  if (threadIdx.x == 0) {
    unsigned int o = 0;
    for (int e = 0; e < E_; ++e) {
      soff[e] = o; offsets[e] = o; cursors[e] = o;
      o += (counts[e] + BM_ - 1) & ~(unsigned)(BM_ - 1);
    }
    soff[E_] = o; offsets[E_] = o;
  }
  __syncthreads();
  for (int r = threadIdx.x; r < RP_; r += blockDim.x) {
    int e = E_ - 1;
    for (int q = 0; q < E_; ++q)
      if ((unsigned)r < soff[q + 1]) { e = q; break; }
    expert_row[r] = e;
  }
}

// ---------------------------------------------------------------------------
// 4) scatter: one block per assignment; grab slot in expert segment, record
//    token/gate, gather x row into bf16 Xg (float4 load -> 4x bf16 pack).
// ---------------------------------------------------------------------------
__global__ void __launch_bounds__(256)
moe_scatter_kernel(const float* __restrict__ x, const int* __restrict__ asg_tok,
                   const int* __restrict__ asg_exp, const float* __restrict__ asg_w,
                   unsigned int* __restrict__ cursors, int* __restrict__ tok_id,
                   float* __restrict__ gate, unsigned short* __restrict__ Xg) {
  __shared__ int spos;
  int a   = blockIdx.x;
  int tok = asg_tok[a];
  int e   = asg_exp[a];
  if (threadIdx.x == 0) {
    unsigned int p = atomicAdd(&cursors[e], 1u);
    spos = (int)p;
    tok_id[p] = tok;
    gate[p]   = asg_w[a];
  }
  __syncthreads();
  int pos = spos;
  const float4* xr = (const float4*)(x + (size_t)tok * H_);
  float4 v = xr[threadIdx.x];                         // 256 * 4 = 1024
  unsigned short h[4] = { f2bf(v.x), f2bf(v.y), f2bf(v.z), f2bf(v.w) };
  unsigned long long packed =
      (unsigned long long)h[0] | ((unsigned long long)h[1] << 16) |
      ((unsigned long long)h[2] << 32) | ((unsigned long long)h[3] << 48);
  *(unsigned long long*)(Xg + (size_t)pos * H_ + threadIdx.x * 4) = packed;
}

// ---------------------------------------------------------------------------
// 5/6) weight convert + transpose: fp32 [mat][R][C] -> bf16 [mat][C][R]
//      so GEMM B-operand is K-contiguous. LDS-tiled 32x32 transpose.
// ---------------------------------------------------------------------------
__global__ void __launch_bounds__(256)
moe_w_transpose_kernel(const float* __restrict__ in, unsigned short* __restrict__ out,
                       int R, int C) {
  __shared__ unsigned short tile[32][33];
  int mat = blockIdx.z;
  int x0 = blockIdx.x * 32, y0 = blockIdx.y * 32;
  int tx = threadIdx.x & 31, ty = threadIdx.x >> 5;   // 32 x 8
  const float* src = in + (size_t)mat * R * C;
  unsigned short* dst = out + (size_t)mat * R * C;
#pragma unroll
  for (int i = 0; i < 4; ++i) {
    int r = y0 + ty + i * 8;
    tile[ty + i * 8][tx] = f2bf(src[(size_t)r * C + x0 + tx]);
  }
  __syncthreads();
#pragma unroll
  for (int i = 0; i < 4; ++i) {
    int orow = x0 + ty + i * 8;
    dst[(size_t)orow * R + y0 + tx] = tile[tx][ty + i * 8];
  }
}

// ---------------------------------------------------------------------------
// WMMA GEMM core: C[128x128] per block, 8 waves, each wave 64x32 (4 M-subtiles
// x 2 N-subtiles = 8 v_wmma_f32_16x16x32_bf16 accumulators). Tiles DMA'd into
// double-buffered LDS by the Tensor Data Mover (wave 0 issues, TENSORcnt
// synchronized), fragments pulled as aligned dword ds_loads matching the
// CDNA5 16-bit A/B VGPR layout (lanes 16-31 take the K=8..15 / K=24..31 half).
// ---------------------------------------------------------------------------
__device__ __forceinline__ FragBF load_frag(const unsigned short* __restrict__ s,
                                            int subBase) {
  int lane = threadIdx.x & 31;
  int r  = subBase + (lane & 15);
  int kb = (lane >> 4) * 8;
  const unsigned short* p = s + r * LDT_ + kb;
  FragBF f;
#pragma unroll
  for (int j = 0; j < 4; ++j) f.u[j]     = *(const unsigned int*)(p + 2 * j);
#pragma unroll
  for (int j = 0; j < 4; ++j) f.u[4 + j] = *(const unsigned int*)(p + 16 + 2 * j);
  return f;
}

template <int EPI>
__global__ void __launch_bounds__(256)
moe_gemm_kernel(const unsigned short* __restrict__ A,   // [RP_][Ktot] bf16
                const unsigned short* __restrict__ BT,  // [E_][Ntot][Ktot] bf16
                const float* __restrict__ bias,         // [E_][Ntot]
                const int* __restrict__ expert_row,
                const unsigned int* __restrict__ offsets,
                int Ktot, int Ntot,
                unsigned short* __restrict__ Hmid,      // EPI==0 output
                const int* __restrict__ tok_id,         // EPI==1
                const float* __restrict__ gate,
                float* __restrict__ out) {
  __shared__ unsigned short sA[2][BM_ * LDT_];
  __shared__ unsigned short sB[2][BN_ * LDT_];

  int rowBase = blockIdx.y * BM_;
  unsigned int total = offsets[E_];
  if ((unsigned)rowBase >= total) return;           // skip unused padded tiles
  int e = expert_row[rowBase];
  int colBase = blockIdx.x * BN_;

  const unsigned short* Abase = A  + (size_t)rowBase * Ktot;
  const unsigned short* Bbase = BT + (size_t)e * Ntot * Ktot + (size_t)colBase * Ktot;

  int wid = threadIdx.x >> 5;
  int wm  = wid & 1;       // row offset 64*wm
  int wn  = wid >> 1;      // col offset 32*wn
  bool isWave0 = (threadIdx.x < 32);

  v8f acc[4][2];
#pragma unroll
  for (int m = 0; m < 4; ++m)
#pragma unroll
    for (int n = 0; n < 2; ++n)
#pragma unroll
      for (int r = 0; r < 8; ++r) acc[m][n][r] = 0.0f;

  int nIter = Ktot / BK_;
  if (isWave0) {                                    // prologue DMA, buffer 0
    tdm_load_tile((unsigned int)(uintptr_t)&sA[0][0], Abase, Ktot);
    tdm_load_tile((unsigned int)(uintptr_t)&sB[0][0], Bbase, Ktot);
  }

  for (int it = 0; it < nIter; ++it) {
    int buf = it & 1;
    if (isWave0) {
      if (it + 1 < nIter) {                         // DMA next tile -> other buf
        int k1 = (it + 1) * BK_;
        tdm_load_tile((unsigned int)(uintptr_t)&sA[buf ^ 1][0], Abase + k1, Ktot);
        tdm_load_tile((unsigned int)(uintptr_t)&sB[buf ^ 1][0], Bbase + k1, Ktot);
        __builtin_amdgcn_s_wait_tensorcnt(2);       // current tile's 2 DMAs done
      } else {
        __builtin_amdgcn_s_wait_tensorcnt(0);
      }
    }
    __syncthreads();                                // tile visible to all waves

    const unsigned short* a = sA[buf];
    const unsigned short* b = sB[buf];
    FragBF bf0 = load_frag(b, wn * 32 + 0);
    FragBF bf1 = load_frag(b, wn * 32 + 16);
    FragBF af0 = load_frag(a, wm * 64 + 0);
    FragBF af1 = load_frag(a, wm * 64 + 16);
    FragBF af2 = load_frag(a, wm * 64 + 32);
    FragBF af3 = load_frag(a, wm * 64 + 48);
    acc[0][0] = __builtin_amdgcn_wmma_f32_16x16x32_bf16(false, af0.v, false, bf0.v, (short)0, acc[0][0], false, false);
    acc[0][1] = __builtin_amdgcn_wmma_f32_16x16x32_bf16(false, af0.v, false, bf1.v, (short)0, acc[0][1], false, false);
    acc[1][0] = __builtin_amdgcn_wmma_f32_16x16x32_bf16(false, af1.v, false, bf0.v, (short)0, acc[1][0], false, false);
    acc[1][1] = __builtin_amdgcn_wmma_f32_16x16x32_bf16(false, af1.v, false, bf1.v, (short)0, acc[1][1], false, false);
    acc[2][0] = __builtin_amdgcn_wmma_f32_16x16x32_bf16(false, af2.v, false, bf0.v, (short)0, acc[2][0], false, false);
    acc[2][1] = __builtin_amdgcn_wmma_f32_16x16x32_bf16(false, af2.v, false, bf1.v, (short)0, acc[2][1], false, false);
    acc[3][0] = __builtin_amdgcn_wmma_f32_16x16x32_bf16(false, af3.v, false, bf0.v, (short)0, acc[3][0], false, false);
    acc[3][1] = __builtin_amdgcn_wmma_f32_16x16x32_bf16(false, af3.v, false, bf1.v, (short)0, acc[3][1], false, false);
    __syncthreads();                                // done reading buf
  }

  // Epilogue. C/D layout: lanes 0-15 -> M = r, lanes 16-31 -> M = 8 + r; N = lane&15.
  int lane = threadIdx.x & 31;
  int cl   = lane & 15;
  int rsel = (lane >> 4) * 8;
#pragma unroll
  for (int m = 0; m < 4; ++m) {
    int rglob = rowBase + wm * 64 + m * 16 + rsel;
#pragma unroll
    for (int n = 0; n < 2; ++n) {
      int cglob = colBase + wn * 32 + n * 16 + cl;
      float bb = bias[e * Ntot + cglob];
      if (EPI == 0) {
#pragma unroll
        for (int r = 0; r < 8; ++r) {
          float h = acc[m][n][r] + bb;
          float g = 0.5f * h * (1.0f + erff(h * 0.70710678118654752f)); // exact GELU
          Hmid[(size_t)(rglob + r) * Ntot + cglob] = f2bf(g);
        }
      } else {
#pragma unroll
        for (int r = 0; r < 8; ++r) {
          int rr  = rglob + r;
          int tok = tok_id[rr];
          if (tok >= 0) {
            float y = acc[m][n][r] + bb;
            atomicAdd(&out[(size_t)tok * Ntot + cglob], gate[rr] * y);
          }
        }
      }
    }
  }
}

// ---------------------------------------------------------------------------
// launch
// ---------------------------------------------------------------------------
extern "C" void kernel_launch(void* const* d_in, const int* in_sizes, int n_in,
                              void* d_out, int out_size, void* d_ws, size_t ws_size,
                              hipStream_t stream) {
  const float* x  = (const float*)d_in[0];   // [T_, H_]
  const float* Wg = (const float*)d_in[1];   // [H_, E_]
  const float* bg = (const float*)d_in[2];   // [E_]
  const float* W1 = (const float*)d_in[3];   // [E_, H_, FF_]
  const float* b1 = (const float*)d_in[4];   // [E_, FF_]
  const float* W2 = (const float*)d_in[5];   // [E_, FF_, H_]
  const float* b2 = (const float*)d_in[6];   // [E_, H_]
  float* out = (float*)d_out;                // [T_, H_]

  char* ws = (char*)d_ws;
  auto alloc = [&](size_t bytes) -> char* {
    char* p = ws;
    ws += (bytes + 255) & ~(size_t)255;
    return p;
  };
  unsigned int* counts   = (unsigned int*)alloc(E_ * 4);
  unsigned int* offsets  = (unsigned int*)alloc((E_ + 1) * 4);
  unsigned int* cursors  = (unsigned int*)alloc(E_ * 4);
  int*   asg_tok    = (int*)  alloc((size_t)A_ * 4);
  int*   asg_exp    = (int*)  alloc((size_t)A_ * 4);
  float* asg_w      = (float*)alloc((size_t)A_ * 4);
  int*   tok_id     = (int*)  alloc((size_t)RP_ * 4);
  float* gate       = (float*)alloc((size_t)RP_ * 4);
  int*   expert_row = (int*)  alloc((size_t)RP_ * 4);
  unsigned short* Xg   = (unsigned short*)alloc((size_t)RP_ * H_ * 2);
  unsigned short* W1T  = (unsigned short*)alloc((size_t)E_ * H_ * FF_ * 2);
  unsigned short* W2T  = (unsigned short*)alloc((size_t)E_ * FF_ * H_ * 2);
  unsigned short* Hmid = (unsigned short*)alloc((size_t)RP_ * FF_ * 2);

  moe_init_kernel<<<512, 256, 0, stream>>>(out, tok_id, counts);
  moe_gate_kernel<<<T_ / 8, 256, 0, stream>>>(x, Wg, bg, counts, asg_tok, asg_exp, asg_w);
  moe_offsets_kernel<<<1, 256, 0, stream>>>(counts, offsets, cursors, expert_row);
  moe_scatter_kernel<<<A_, 256, 0, stream>>>(x, asg_tok, asg_exp, asg_w, cursors,
                                             tok_id, gate, Xg);
  // W1 [E][H][FF] -> W1T [E][FF][H];  W2 [E][FF][H] -> W2T [E][H][FF]
  moe_w_transpose_kernel<<<dim3(FF_ / 32, H_ / 32, E_), 256, 0, stream>>>(W1, W1T, H_, FF_);
  moe_w_transpose_kernel<<<dim3(H_ / 32, FF_ / 32, E_), 256, 0, stream>>>(W2, W2T, FF_, H_);

  // GEMM1: Hmid = gelu(Xg @ W1[e] + b1[e])   [rows x FF_], K = H_
  moe_gemm_kernel<0><<<dim3(FF_ / BN_, RP_ / BM_), 256, 0, stream>>>(
      Xg, W1T, b1, expert_row, offsets, H_, FF_, Hmid, nullptr, nullptr, nullptr);
  // GEMM2: out += gate * (Hmid @ W2[e] + b2[e])   [rows x H_], K = FF_
  moe_gemm_kernel<1><<<dim3(H_ / BN_, RP_ / BM_), 256, 0, stream>>>(
      Hmid, W2T, b2, expert_row, offsets, FF_, H_, nullptr, tok_id, gate, out);
}
